// RecurrentDecoder_87917980549847
// MI455X (gfx1250) — compile-verified, matrix operand order checked by reading, not run on hardware
//
#include <hip/hip_runtime.h>
#include <hip/hip_bf16.h>
#include <stdint.h>

// ---------------- problem constants ----------------
#define Bq 512
#define Tq 512
#define Fq 128
#define Lq 512

// WMMA fragment geometry: 16x16x32 bf16, B-fragment = 32x16 bf16
// stored lane-major: dword index = lane*8 + v  (256 dwords / fragment)
#define FRAG_DW 256
#define KT_F   (Fq/32)            // 4  k-tiles from xt
#define KT_L   (Lq/32)            // 16 k-tiles from h
#define KT_C   (KT_F + KT_L)      // 20 combined k-tiles per gate
#define NLT    (Lq/16)            // 32 latent N-tiles
// gate fragments: (g in r,z,n) x (lt 0..31) x (kt 0..19); kt<4 from W_ih, kt>=4 from W_hh
#define GATE_DW (3*NLT*KT_C*FRAG_DW)            // 491520 dw
#define WD_OFF  GATE_DW
#define TOTAL_DW (WD_OFF + (Fq/16)*KT_L*FRAG_DW) // 524288 dw = 2 MiB

typedef __bf16 v16bf __attribute__((ext_vector_type(16)));
typedef float  v8f  __attribute__((ext_vector_type(8)));

union Frag { v16bf v; float4 f[2]; };

__device__ __forceinline__ int gfrag(int g, int lt, int kt) {
  return ((g * NLT + lt) * KT_C + kt) * FRAG_DW;
}

__device__ __forceinline__ v8f wmma_bf16(v16bf a, v16bf b, v8f c) {
  return __builtin_amdgcn_wmma_f32_16x16x32_bf16(
      /*neg_a=*/false, a, /*neg_b=*/false, b,
      /*c_mod=*/(short)0, c, /*reuse_a=*/false, /*reuse_b=*/false);
}

__device__ __forceinline__ float sigm_(float x) { return 1.0f / (1.0f + __expf(-x)); }
__device__ __forceinline__ float tanh_(float x) { return 2.0f / (1.0f + __expf(-2.0f * x)) - 1.0f; }

// A-fragment (16x32 bf16, 16-bit A layout) rebuilt from row-major bf16 LDS.
// lane l: M = l&15 ; VGPR v packs (K,K+1); lanes 16-31 carry the +8 K offset.
__device__ __forceinline__ v16bf loadA(const uint32_t* lds, int rowStrideDw, int kt, int lane) {
  int row   = lane & 15;
  int dbase = row * rowStrideDw + kt * 16 + ((lane & 16) ? 4 : 0);
  Frag u;
  u.f[0] = *(const float4*)(lds + dbase);      // K pairs 0..7  (or 8..15)
  u.f[1] = *(const float4*)(lds + dbase + 8);  // K pairs 16..23 (or 24..31)
  return u.v;
}

// B-fragment: 32 contiguous bytes per lane -> two global b128 loads, L2-resident.
__device__ __forceinline__ v16bf loadB(const uint32_t* frags, int fragDwOff, int lane) {
  Frag u;
  const float4* p = (const float4*)(frags + fragDwOff + lane * 8);
  u.f[0] = p[0];
  u.f[1] = p[1];
  return u.v;
}

// ---------------- weight pre-pack: fp32 row-major -> bf16 B-fragments ----------------
// Gate region: combined K=[F | L] streams; B(k,n) = W[n,k].
__global__ void prep_weights(const float* __restrict__ Wih,   // [3L, F]
                             const float* __restrict__ Whh,   // [3L, L]
                             const float* __restrict__ Wd,    // [F, L]
                             uint32_t* __restrict__ frags) {
  int id = blockIdx.x * 256 + threadIdx.x;
  if (id >= TOTAL_DW) return;

  int rel, f, r, lane, v, kk;
  float a, b;
  if (id < WD_OFF) {
    rel  = id;
    f    = rel >> 8;            // fragment index
    r    = rel & 255;
    lane = r >> 3; v = r & 7;
    int kt  = f % KT_C;
    int ltg = f / KT_C;
    int lt  = ltg % NLT;
    int g   = ltg / NLT;
    int n   = g * Lq + lt * 16 + (lane & 15);   // row in stacked [3L] dim
    kk = (v < 4 ? 2 * v : 16 + 2 * (v - 4)) + ((lane & 16) ? 8 : 0);
    if (kt < KT_F) {
      int K = kt * 32 + kk;
      a = Wih[n * Fq + K];  b = Wih[n * Fq + K + 1];
    } else {
      int K = (kt - KT_F) * 32 + kk;
      a = Whh[n * Lq + K];  b = Whh[n * Lq + K + 1];
    }
  } else {
    rel  = id - WD_OFF;
    f    = rel >> 8;
    r    = rel & 255;
    lane = r >> 3; v = r & 7;
    int nt = f / KT_L, kt = f % KT_L;
    int n  = nt * 16 + (lane & 15);
    kk = (v < 4 ? 2 * v : 16 + 2 * (v - 4)) + ((lane & 16) ? 8 : 0);
    int K = kt * 32 + kk;
    a = Wd[n * Lq + K];  b = Wd[n * Lq + K + 1];
  }
  union { __bf16 h[2]; uint32_t u; } pk;
  pk.h[0] = (__bf16)a;
  pk.h[1] = (__bf16)b;
  frags[id] = pk.u;
}

// ---------------- fused GRU decoder: one 16-row batch tile per WG ----------------
__global__ __launch_bounds__(256) void gru_decoder(
    const float* __restrict__ h0,    // [1,B,L]
    const float* __restrict__ b_ih,  // [3L]
    const float* __restrict__ b_hh,  // [3L]
    const float* __restrict__ bd,    // [F]
    const uint32_t* __restrict__ frags,
    float* __restrict__ out)         // [B,T,F]
{
  __shared__ __align__(16) __bf16 hB[2][16 * Lq];  // 32 KB, double-buffered h
  __shared__ __align__(16) __bf16 xB[2][16 * Fq];  //  8 KB, double-buffered xt

  const int tid   = threadIdx.x;
  const int wave  = tid >> 5;
  const int lane  = tid & 31;
  const int bTile = blockIdx.x;
  const int cL    = lane & 15;
  const int mOff  = (lane & 16) ? 8 : 0;   // C-layout: VGPR e -> row e (+8 for hi lanes)

  // per-lane biases for this wave's 4 latent tiles; r/z biases pre-summed (i+h)
  float bR[4], bZ[4], biN[4], bhN[4];
#pragma unroll
  for (int i = 0; i < 4; i++) {
    int col = (wave * 4 + i) * 16 + cL;
    bR[i]  = b_ih[col]          + b_hh[col];
    bZ[i]  = b_ih[Lq + col]     + b_hh[Lq + col];
    biN[i] = b_ih[2 * Lq + col];
    bhN[i] = b_hh[2 * Lq + col];
  }
  const float bdv = bd[wave * 16 + cL];

  // stage h0 tile into LDS (bf16)
  for (int e = tid; e < 16 * Lq; e += 256) {
    int m = e >> 9, l = e & (Lq - 1);
    hB[0][e] = (__bf16)h0[(size_t)(bTile * 16 + m) * Lq + l];
  }
  __syncthreads();

  // initial xt = dense(h0) + bd  (not emitted to out)
  {
    v8f acc = {0.f, 0.f, 0.f, 0.f, 0.f, 0.f, 0.f, 0.f};
#pragma unroll 4
    for (int kt = 0; kt < KT_L; kt++) {
      v16bf a = loadA((const uint32_t*)&hB[0][0], Lq / 2, kt, lane);
      acc = wmma_bf16(a, loadB(frags, WD_OFF + (wave * KT_L + kt) * FRAG_DW, lane), acc);
    }
    int colF = wave * 16 + cL;
#pragma unroll
    for (int e = 0; e < 8; e++)
      xB[0][(e + mOff) * Fq + colF] = (__bf16)(acc[e] + bdv);
  }
  __syncthreads();

  // -------- sequential recurrence, all in-WG --------
  for (int t = 0; t < Tq; t++) {
    const int p = t & 1, q = p ^ 1;
    const uint32_t* xLds = (const uint32_t*)&xB[p][0];
    const uint32_t* hLds = (const uint32_t*)&hB[p][0];

    // gates for this wave's 4 latent tiles
    for (int i = 0; i < 4; i++) {
      const int lt = wave * 4 + i;

      // r and z gates: i_g and h_g share one accumulator each (only their sum is used)
      v8f ar = {0.f,0.f,0.f,0.f,0.f,0.f,0.f,0.f};
      v8f az = ar;
#pragma unroll
      for (int kt = 0; kt < KT_F; kt++) {
        v16bf a = loadA(xLds, Fq / 2, kt, lane);
        ar = wmma_bf16(a, loadB(frags, gfrag(0, lt, kt), lane), ar);
        az = wmma_bf16(a, loadB(frags, gfrag(1, lt, kt), lane), az);
      }
#pragma unroll 4
      for (int kt = KT_F; kt < KT_C; kt++) {
        __builtin_prefetch(frags + gfrag(2, lt, kt), 0, 0);
        v16bf a = loadA(hLds, Lq / 2, kt - KT_F, lane);
        ar = wmma_bf16(a, loadB(frags, gfrag(0, lt, kt), lane), ar);
        az = wmma_bf16(a, loadB(frags, gfrag(1, lt, kt), lane), az);
      }

      // n gate: i_n and h_n must stay separate (r scales h_n)
      v8f an = {0.f,0.f,0.f,0.f,0.f,0.f,0.f,0.f};
#pragma unroll
      for (int kt = 0; kt < KT_F; kt++) {
        v16bf a = loadA(xLds, Fq / 2, kt, lane);
        an = wmma_bf16(a, loadB(frags, gfrag(2, lt, kt), lane), an);
      }
      v8f ahn = {0.f,0.f,0.f,0.f,0.f,0.f,0.f,0.f};
#pragma unroll 4
      for (int kt = KT_F; kt < KT_C; kt++) {
        v16bf a = loadA(hLds, Lq / 2, kt - KT_F, lane);
        ahn = wmma_bf16(a, loadB(frags, gfrag(2, lt, kt), lane), ahn);
      }

      // GRU elementwise; C layout: VGPR e holds (M = e+mOff, N = lane&15)
      const int col = lt * 16 + cL;
#pragma unroll
      for (int e = 0; e < 8; e++) {
        int m    = e + mOff;
        float r  = sigm_(ar[e] + bR[i]);
        float z  = sigm_(az[e] + bZ[i]);
        float nn = tanh_(an[e] + biN[i] + r * (ahn[e] + bhN[i]));
        float ho = (float)hB[p][m * Lq + col];
        hB[q][m * Lq + col] = (__bf16)((1.0f - z) * nn + z * ho);
      }
    }
    __syncthreads();

    // x_new = h_new * Wd^T + bd ; feed next step + emit out[:, t]
    {
      v8f acc = {0.f, 0.f, 0.f, 0.f, 0.f, 0.f, 0.f, 0.f};
#pragma unroll 4
      for (int kt = 0; kt < KT_L; kt++) {
        v16bf a = loadA((const uint32_t*)&hB[q][0], Lq / 2, kt, lane);
        acc = wmma_bf16(a, loadB(frags, WD_OFF + (wave * KT_L + kt) * FRAG_DW, lane), acc);
      }
      const int colF = wave * 16 + cL;
#pragma unroll
      for (int e = 0; e < 8; e++) {
        int m    = e + mOff;
        float v  = acc[e] + bdv;
        xB[q][m * Fq + colF] = (__bf16)v;
        out[((size_t)(bTile * 16 + m) * Tq + t) * Fq + colF] = v;
      }
    }
    __syncthreads();
  }
}

extern "C" void kernel_launch(void* const* d_in, const int* in_sizes, int n_in,
                              void* d_out, int out_size, void* d_ws, size_t ws_size,
                              hipStream_t stream) {
  // setup_inputs order: x, h_0, W_ih, W_hh, b_ih, b_hh, Wd, bd, seq_len
  // x (d_in[0]) is never read: with epsilon=0 the model is pure closed-loop decode.
  const float* h0  = (const float*)d_in[1];
  const float* Wih = (const float*)d_in[2];
  const float* Whh = (const float*)d_in[3];
  const float* bih = (const float*)d_in[4];
  const float* bhh = (const float*)d_in[5];
  const float* Wd  = (const float*)d_in[6];
  const float* bd  = (const float*)d_in[7];
  (void)in_sizes; (void)n_in; (void)out_size; (void)ws_size;

  uint32_t* frags = (uint32_t*)d_ws;   // 2 MiB of bf16 WMMA B-fragments

  prep_weights<<<TOTAL_DW / 256, 256, 0, stream>>>(Wih, Whh, Wd, frags);
  gru_decoder<<<Bq / 16, 256, 0, stream>>>(h0, bih, bhh, bd, frags, (float*)d_out);
}
